// CrossLayer_1348619731627
// MI455X (gfx1250) — matrix operand order repaired and checked by reading, not compiled, not run on hardware
//
#include <hip/hip_runtime.h>

#define B_ 8
#define C_ 256
#define S_ 2048
#define D_ 64
#define NW 4           // waves per block
#define CH 64          // keys per stream chunk
#define VSTR 72        // ushorts per V row in LDS: 128B data + 16B TDM pad = 144B (16B aligned, conflict-free)
#define KSTR 72        // ushorts per K row in LDS: 128B data + 16B TDM pad = 144B
#define PSTR 72        // ushorts per P row in LDS (16 rows x 64 K + pad)

typedef __attribute__((ext_vector_type(16))) __bf16          v16bf;
typedef __attribute__((ext_vector_type(16))) unsigned short  v16u;
typedef __attribute__((ext_vector_type(8)))  float           v8f;
typedef __attribute__((ext_vector_type(4)))  unsigned int    v4u;
typedef __attribute__((ext_vector_type(8)))  int             v8i;
typedef __attribute__((ext_vector_type(4)))  int             v4i;

union TileU {
    v16u  u;
    v16bf b;
    unsigned short s[16];
    unsigned int   w32[8];
    uint4 q[2];
};

__device__ __forceinline__ unsigned short f32_bf16_rne(float f) {
    union { float f; unsigned int u; } x; x.f = f;
    unsigned int r = x.u + 0x7FFFu + ((x.u >> 16) & 1u);
    return (unsigned short)(r >> 16);
}

// XOR-lane shuffle within 16-lane DPP rows: dpp_ctrl = ROW_XMASK(m) = 0x160|m.
// Pure VALU (v_mov_b32_dpp) -- no LDS traffic, no DScnt waits.
template <int CTRL>
__device__ __forceinline__ float dpp_xor(float x) {
    return __int_as_float(__builtin_amdgcn_update_dpp(
        0, __float_as_int(x), CTRL, 0xF, 0xF, true));
}

// ---------------------------------------------------------------------------
// Pre-pass: f32 -> bf16 (RNE), vectorized float4 -> ushort4. Memory bound.
// ---------------------------------------------------------------------------
__global__ void cvt_f32_bf16(const float* __restrict__ src,
                             unsigned short* __restrict__ dst, int n4) {
    int i = blockIdx.x * blockDim.x + threadIdx.x;
    if (i >= n4) return;
    float4 v = ((const float4*)src)[i];
    ushort4 o;
    o.x = f32_bf16_rne(v.x); o.y = f32_bf16_rne(v.y);
    o.z = f32_bf16_rne(v.z); o.w = f32_bf16_rne(v.w);
    ((ushort4*)dst)[i] = o;
}

// ---------------------------------------------------------------------------
// Build a 2D TDM descriptor (D#) and issue TENSOR_LOAD_TO_LDS.
// bf16 elements (data_size=2B); pad_enable inserts (pad_amt+1) dwords of LDS
// padding after every 2^(pad_int+1) dwords, giving bank-conflict-free rows.
// ---------------------------------------------------------------------------
__device__ __forceinline__ void tdm_load_2d(unsigned int lds_addr,
                                            unsigned long long gaddr,
                                            unsigned int dim0, unsigned int dim1,
                                            unsigned int stride0,
                                            unsigned int tile0, unsigned int tile1,
                                            unsigned int pad_int, unsigned int pad_amt) {
    v4u g0;
    g0[0] = 1u;                                               // count=1, user mode, no gather
    g0[1] = lds_addr;                                         // LDS byte address
    g0[2] = (unsigned int)(gaddr & 0xFFFFFFFFu);              // global addr [31:0]
    g0[3] = (unsigned int)((gaddr >> 32) & 0x01FFFFFFu)       // global addr [56:32]
          | (2u << 30);                                       // type = 2 ("image")
    v8i g1;
    g1[0] = (int)((1u << 16)                                  // data_size = 2 bytes
                | (1u << 20)                                  // pad_enable
                | (pad_int << 22) | (pad_amt << 25));
    g1[1] = (int)((dim0 & 0xFFFFu) << 16);                    // tensor_dim0 lo16 (atomic_barrier=0)
    g1[2] = (int)(((dim0 >> 16) & 0xFFFFu) | ((dim1 & 0xFFFFu) << 16));
    g1[3] = (int)(((dim1 >> 16) & 0xFFFFu) | ((tile0 & 0xFFFFu) << 16));
    g1[4] = (int)(tile1 & 0xFFFFu);                           // tile_dim1 (tile_dim2 = 0)
    g1[5] = (int)stride0;                                     // tensor_dim0_stride lo32
    g1[6] = 0;                                                // stride0 hi16 | stride1 lo16
    g1[7] = 0;
    v4i g2 = (v4i)0;
    v4i g3 = (v4i)0;
    v8i g4 = (v8i)0;                                          // extra group (clang-23 6-arg form)
    __builtin_amdgcn_tensor_load_to_lds(g0, g1, g2, g3, g4, 0);
}

// ---------------------------------------------------------------------------
// Fused flash-attention-style kernel:
//   outT = softmax(b @ c^T) @ a^T  (per batch), out = that^T + a
// One wave owns 16 query rows; 64-key chunks stream via TDM, double buffered.
// ---------------------------------------------------------------------------
__global__ __launch_bounds__(NW * 32, 1)
void fused_attn_tdm(const float* __restrict__ a,           // [B, C, S]  f32 (residual)
                    float* __restrict__ out,               // [B, C, S]  f32
                    const unsigned short* __restrict__ wa, // a  bf16
                    const unsigned short* __restrict__ wb, // b  bf16 ("Q")
                    const unsigned short* __restrict__ wc) // c  bf16 ("K")
{
    __shared__ unsigned short ldsV[2][C_ * VSTR];   // 2 x 36 KB  (a slab: 256 rows x 64 m)
    __shared__ unsigned short ldsK[2][CH * KSTR];   // 2 x 9 KB  (c slab: 64 rows x 64 d)
    __shared__ unsigned short ldsP[NW][16 * PSTR];  // 9 KB  P staging (C-layout -> A-layout)

    const int tid    = threadIdx.x;
    const int lane   = tid & 31;
    const int wave   = tid >> 5;
    const int laneLo = lane & 15;
    const int hi     = lane >> 4;
    const int hi8    = hi * 8;
    const int hi16   = hi * 16;

    const int batch = blockIdx.y;
    const int n0    = blockIdx.x * (16 * NW) + wave * 16;

    const float*          aB = a  + (size_t)batch * C_ * S_;
    const unsigned short* qB = wb + (size_t)batch * S_ * D_;

    // ---- Q tile (16 x 64) -> two bf16 A-fragments; data already bf16, pairs packed ----
    TileU qa[2];
#pragma unroll
    for (int w = 0; w < 2; ++w) {
#pragma unroll
        for (int v = 0; v < 8; ++v) {
            int kk = 2 * v;
            int d  = w * 32 + kk + hi16;
            qa[w].w32[v] = *(const unsigned int*)(qB + (size_t)(n0 + laneLo) * D_ + d);
        }
    }

    v8f acc[16];
#pragma unroll
    for (int t = 0; t < 16; ++t) acc[t] = (v8f)0.0f;
    float mrun[8], lrun[8];
#pragma unroll
    for (int r = 0; r < 8; ++r) { mrun[r] = -1e30f; lrun[r] = 0.0f; }

    const unsigned long long aG =
        (unsigned long long)(uintptr_t)(wa + (size_t)batch * C_ * S_);
    const unsigned long long kG =
        (unsigned long long)(uintptr_t)(wc + (size_t)batch * S_ * D_);

    // prologue: TDM chunk 0 into buffer 0 (wave 0 drives the DMA engine)
    if (wave == 0) {
        tdm_load_2d((unsigned int)(uintptr_t)&ldsV[0][0], aG,
                    S_, C_, S_, CH, C_, /*pad: 128B rows +16B*/ 4, 3);
        tdm_load_2d((unsigned int)(uintptr_t)&ldsK[0][0], kG,
                    D_, S_, D_, D_, CH, 4, 3);
    }

    const int NCHUNK = S_ / CH;
    for (int it = 0; it < NCHUNK; ++it) {
        const int buf = it & 1;
        if (wave == 0) __builtin_amdgcn_s_wait_tensorcnt(0);
        __syncthreads();   // slab `buf` visible to all waves; prev compute done
        if (wave == 0 && (it + 1) < NCHUNK) {
            const unsigned long long m1 = (unsigned long long)(it + 1) * CH;
            const int nb = buf ^ 1;
            tdm_load_2d((unsigned int)(uintptr_t)&ldsV[nb][0], aG + m1 * 2,
                        S_, C_, S_, CH, C_, 4, 3);
            tdm_load_2d((unsigned int)(uintptr_t)&ldsK[nb][0], kG + m1 * D_ * 2,
                        D_, S_, D_, D_, CH, 4, 3);
        }

        // ---- scores: four 16x16 tiles, each a 2-deep WMMA chain over d ----
        v8f s[4];
#pragma unroll
        for (int tt = 0; tt < 4; ++tt) {
            TileU kf0, kf1;
            const unsigned short* kr = &ldsK[buf][(tt * 16 + laneLo) * KSTR + hi16];
            kf0.q[0] = *(const uint4*)(kr);
            kf0.q[1] = *(const uint4*)(kr + 8);
            kf1.q[0] = *(const uint4*)(kr + 32);
            kf1.q[1] = *(const uint4*)(kr + 40);
            s[tt] = __builtin_amdgcn_wmma_f32_16x16x32_bf16(
                false, qa[0].b, false, kf0.b, (short)0, (v8f)0.0f, false, false);
            s[tt] = __builtin_amdgcn_wmma_f32_16x16x32_bf16(
                false, qa[1].b, false, kf1.b, (short)0, s[tt], false, false);
        }

        // ---- online softmax over 64 keys; DPP ROW_XMASK reductions ----
        float sc[8];
#pragma unroll
        for (int r = 0; r < 8; ++r) {
            float cm = fmaxf(fmaxf(s[0][r], s[1][r]), fmaxf(s[2][r], s[3][r]));
            cm = fmaxf(cm, dpp_xor<0x161>(cm));   // xor 1
            cm = fmaxf(cm, dpp_xor<0x162>(cm));   // xor 2
            cm = fmaxf(cm, dpp_xor<0x164>(cm));   // xor 4
            cm = fmaxf(cm, dpp_xor<0x168>(cm));   // xor 8
            float mnew = fmaxf(mrun[r], cm);
            sc[r]   = __expf(mrun[r] - mnew);
            mrun[r] = mnew;
            float p0 = __expf(s[0][r] - mnew);
            float p1 = __expf(s[1][r] - mnew);
            float p2 = __expf(s[2][r] - mnew);
            float p3 = __expf(s[3][r] - mnew);
            s[0][r] = p0; s[1][r] = p1; s[2][r] = p2; s[3][r] = p3;
            float rs = (p0 + p1) + (p2 + p3);
            rs += dpp_xor<0x161>(rs);
            rs += dpp_xor<0x162>(rs);
            rs += dpp_xor<0x164>(rs);
            rs += dpp_xor<0x168>(rs);
            lrun[r] = lrun[r] * sc[r] + rs;
        }
        v8f scv;
#pragma unroll
        for (int r = 0; r < 8; ++r) scv[r] = sc[r];
#pragma unroll
        for (int t = 0; t < 16; ++t) acc[t] *= scv;   // packs into v_pk_mul_f32

        // ---- stage P (C-layout f32) -> LDS bf16 [M][K]; read back in A layout ----
        unsigned short* myP = &ldsP[wave][0];
#pragma unroll
        for (int r = 0; r < 8; ++r) {
            int row = (r + hi8) * PSTR;
#pragma unroll
            for (int i = 0; i < 4; ++i)
                myP[row + i * 16 + laneLo] =
                    (unsigned short)((__float_as_uint(s[i][r]) + 0x8000u) >> 16);
        }
        TileU pa[2];   // same-wave DS ops are processed in order by the LDS unit
        pa[0].q[0] = *(const uint4*)&myP[laneLo * PSTR + hi16];       // K  0..31
        pa[0].q[1] = *(const uint4*)&myP[laneLo * PSTR + 8 + hi16];
        pa[1].q[0] = *(const uint4*)&myP[laneLo * PSTR + 32 + hi16];  // K 32..63
        pa[1].q[1] = *(const uint4*)&myP[laneLo * PSTR + 40 + hi16];

        // ---- PV: acc[n, c] += P[16x64] @ V[64x16]; rotating V-fragment prefetch ----
        TileU vb[2];
        {
            const unsigned short* vrow0 = &ldsV[buf][laneLo * VSTR + hi16];
            vb[0].q[0] = *(const uint4*)(vrow0);
            vb[0].q[1] = *(const uint4*)(vrow0 + 8);
        }
#pragma unroll
        for (int u = 0; u < 32; ++u) {          // u = t*2 + h (c-tile t, K-half h)
            if (u < 31) {                       // prefetch next fragment under this wmma
                const int un = u + 1;
                const unsigned short* vrow =
                    &ldsV[buf][((un >> 1) * 16 + laneLo) * VSTR + (un & 1) * 32 + hi16];
                vb[un & 1].q[0] = *(const uint4*)(vrow);
                vb[un & 1].q[1] = *(const uint4*)(vrow + 8);
            }
            acc[u >> 1] = __builtin_amdgcn_wmma_f32_16x16x32_bf16(
                false, pa[u & 1].b, false, vb[u & 1].b, (short)0, acc[u >> 1], false, false);
        }
    }

    // ---- epilogue: out[b, c, n] = acc[n, c] / l[n] + a[b, c, n] (f32 residual) ----
    float inv[8];
#pragma unroll
    for (int r = 0; r < 8; ++r) inv[r] = 1.0f / lrun[r];
    float* oB = out + (size_t)batch * C_ * S_;
#pragma unroll
    for (int t = 0; t < 16; ++t) {
        int cg = t * 16 + laneLo;
#pragma unroll
        for (int r = 0; r < 8; ++r) {
            size_t idx = (size_t)cg * S_ + (size_t)(n0 + r + hi8);
            oB[idx] = acc[t][r] * inv[r] + aB[idx];
        }
    }
}

extern "C" void kernel_launch(void* const* d_in, const int* in_sizes, int n_in,
                              void* d_out, int out_size, void* d_ws, size_t ws_size,
                              hipStream_t stream) {
    const float* a  = (const float*)d_in[0];
    const float* bq = (const float*)d_in[1];
    const float* ck = (const float*)d_in[2];
    float* out = (float*)d_out;

    // workspace layout (bf16 copies): a: 8 MiB | b: 2 MiB | c: 2 MiB  (needs ~12.6 MB)
    unsigned short* wa = (unsigned short*)d_ws;
    unsigned short* wb = wa + (size_t)B_ * C_ * S_;
    unsigned short* wc = wb + (size_t)B_ * S_ * D_;

    const int n4a = B_ * C_ * S_ / 4;
    const int n4b = B_ * S_ * D_ / 4;
    cvt_f32_bf16<<<(n4a + 255) / 256, 256, 0, stream>>>(a,  wa, n4a);
    cvt_f32_bf16<<<(n4b + 255) / 256, 256, 0, stream>>>(bq, wb, n4b);
    cvt_f32_bf16<<<(n4b + 255) / 256, 256, 0, stream>>>(ck, wc, n4b);

    dim3 grid(S_ / (16 * NW), B_);   // (32, 8)
    fused_attn_tdm<<<grid, NW * 32, 0, stream>>>(a, out, wa, wb, wc);
}
